// GeometricusPerc_85074712199233
// MI455X (gfx1250) — matrix operand order repaired and checked by reading, not compiled, not run on hardware
//
#include <hip/hip_runtime.h>

#define D 16
#define H 33
#define HPAD_K 64
#define K_NB 16
#define N_ATOM 6000
#define N_SURF 12000
#define N_MOL 6
#define APM (N_ATOM / N_MOL) /* 1000 atoms per molecule  */
#define SPM (N_SURF / N_MOL) /* 2000 surf pts per mol    */
#define EPSV 1e-5f
#define SLOPE 0.2f

typedef __attribute__((ext_vector_type(16))) _Float16 v16h;
typedef __attribute__((ext_vector_type(8)))  _Float16 v8h;
typedef __attribute__((ext_vector_type(8)))  float    v8f;

__device__ __forceinline__ float leaky(float x) { return x >= 0.f ? x : SLOPE * x; }

__device__ __forceinline__ unsigned int pack2(float a, float b) {
  union { _Float16 h[2]; unsigned int u; } pk;
  pk.h[0] = (_Float16)a; pk.h[1] = (_Float16)b;
  return pk.u;
}

// ---- A-fragment: per ISA layout lane L holds row M=L%16; its 16 halfs are two
//      contiguous 8-half runs at K = kOff+8*(L>>4) and +16  ->  2 x b128 loads ----
__device__ __forceinline__ v16h load_a_frag(const _Float16* base, int ldk, int kOff) {
  const int lane = threadIdx.x & 31;
  const _Float16* p = base + (lane & 15) * ldk + kOff + 8 * (lane >> 4);
  union { v16h v; v8h h[2]; } u;
  u.h[0] = *(const v8h*)(p);
  u.h[1] = *(const v8h*)(p + 16);
  return u.v;
}

// ---- B-fragment from pre-swizzled weights: lane's 16 halfs are 32 contiguous
//      bytes at ws[(fragId*32 + lane)*16]  ->  one 32B vector load ----
__device__ __forceinline__ v16h load_b_fragS(const _Float16* ws, int fragId) {
  const int lane = threadIdx.x & 31;
  return *(const v16h*)(ws + (fragId * 32 + lane) * 16);
}

// ---- one-shot: convert + zero-pad + swizzle all weights into WMMA B-fragment
//      order (lane L of frag (kS,nT) holds w[kS*32+8*(L>>4)+(j&7)+16*(j>>3)][nT*16+(L&15)]) ----
__global__ void prep_weights(const float* tt_w1, const float* tt_w2,
                             const float* aa_w1, const float* se_w1,
                             const float* aa_w2, const float* se_w2,
                             _Float16* tt1s /*[32][16]*/, _Float16* tt2s /*[32][16]*/,
                             _Float16* w1s /*[6][6][32][16]*/, _Float16* w2s /*[6][2][32][16]*/) {
  const int t0 = blockIdx.x * blockDim.x + threadIdx.x;
  const int stride = gridDim.x * blockDim.x;
  for (int i = t0; i < 512; i += stride) {
    const int L = i >> 4, j = i & 15;
    const int k = 8 * (L >> 4) + (j & 7) + 16 * (j >> 3);
    const int n = L & 15;
    tt1s[i] = (_Float16)((k < 16) ? tt_w1[k * 16 + n] : 0.f);
    tt2s[i] = (_Float16)((k < 16) ? tt_w2[k * 16 + n] : 0.f);
  }
  for (int i = t0; i < 6 * 6 * 512; i += stride) {
    const int l = i / (6 * 512);
    const int rem = i % (6 * 512);
    const int f = rem >> 9, t = rem & 511;
    const int L = t >> 4, j = t & 15;
    const int kS = f / 3, nT = f % 3;
    const int k = kS * 32 + 8 * (L >> 4) + (j & 7) + 16 * (j >> 3);
    const int n = nT * 16 + (L & 15);
    const float* s1 = (l < 3) ? (aa_w1 + l * H * H) : (se_w1 + (l - 3) * H * H);
    w1s[i] = (_Float16)((k < H && n < H) ? s1[k * H + n] : 0.f);
  }
  for (int i = t0; i < 6 * 2 * 512; i += stride) {
    const int l = i / (2 * 512);
    const int rem = i % (2 * 512);
    const int kS = rem >> 9, t = rem & 511;
    const int L = t >> 4, j = t & 15;
    const int k = kS * 32 + 8 * (L >> 4) + (j & 7) + 16 * (j >> 3);
    const int n = L & 15;
    const float* s2 = (l < 3) ? (aa_w2 + l * H * D) : (se_w2 + (l - 3) * H * D);
    w2s[i] = (_Float16)((k < H) ? s2[k * D + n] : 0.f);
  }
}

// ---- atom-type MLP: leaky(X@W1+b1)@W2+b2, 16-row tile per wave, 2 WMMAs ----
__global__ void __launch_bounds__(32) tt_kernel(const float* atomtypes, const float* tt_b1,
                                                const float* tt_b2, const _Float16* tt1s,
                                                const _Float16* tt2s, float* outA) {
  __shared__ __align__(16) _Float16 Als[16 * 32];
  __shared__ __align__(16) _Float16 Hls[16 * 32];
  const int tile = blockIdx.x;
  const int lane = threadIdx.x;
  for (int t = lane; t < 16 * 16; t += 32) {         // packed pair stores
    const int r = t >> 4, cp = t & 15, col = cp * 2;
    unsigned int u = 0u;
    if (cp < 8) {
      const float* src = atomtypes + (tile * 16 + r) * 16 + col;
      u = pack2(src[0], src[1]);
    }
    *(unsigned int*)(Als + r * 32 + col) = u;
  }
  __syncthreads();
  v8f c = {};
  {
    v16h a = load_a_frag(Als, 32, 0);
    v16h b = load_b_fragS(tt1s, 0);
    c = __builtin_amdgcn_wmma_f32_16x16x32_f16(false, a, false, b, (short)0, c, false, false);
  }
  const int col = lane & 15, hh = lane >> 4;
  const float bias1 = tt_b1[col];
#pragma unroll
  for (int r = 0; r < 8; ++r)
    Hls[(r + 8 * hh) * 32 + col] = (_Float16)leaky(c[r] + bias1);
  for (int t = lane; t < 16 * 8; t += 32)            // zero K=16..31 (packed)
    *(unsigned int*)(Hls + (t >> 3) * 32 + 16 + 2 * (t & 7)) = 0u;
  __syncthreads();
  v8f c2 = {};
  {
    v16h a = load_a_frag(Hls, 32, 0);
    v16h b = load_b_fragS(tt2s, 0);
    c2 = __builtin_amdgcn_wmma_f32_16x16x32_f16(false, a, false, b, (short)0, c2, false, false);
  }
  const float bias2 = tt_b2[col];
#pragma unroll
  for (int r = 0; r < 8; ++r)
    outA[(tile * 16 + r + 8 * hh) * 16 + col] = c2[r] + bias2;
}

// ---- per-molecule brute-force KNN, top-16 (self excluded for atom-atom) ----
__global__ void knn_kernel(const float* q, const float* c, int nq, int perMolQ, int perMolC,
                           int excludeSelf, int* idxOut, float* dOut) {
  const int i = blockIdx.x * blockDim.x + threadIdx.x;
  if (i >= nq) return;
  const int base = (i / perMolQ) * perMolC;
  const float qx = q[3 * i], qy = q[3 * i + 1], qz = q[3 * i + 2];
  float bd[K_NB];
  int bi[K_NB];
#pragma unroll
  for (int k = 0; k < K_NB; ++k) { bd[k] = 3.4e38f; bi[k] = base; }
  for (int j = 0; j < perMolC; ++j) {
    const int cj = base + j;
    if (excludeSelf && cj == i) continue;
    const float dx = qx - c[3 * cj], dy = qy - c[3 * cj + 1], dz = qz - c[3 * cj + 2];
    const float d2 = dx * dx + dy * dy + dz * dz;
    if (d2 < bd[K_NB - 1]) {
      int p = K_NB - 1;
      while (p > 0 && bd[p - 1] > d2) { bd[p] = bd[p - 1]; bi[p] = bi[p - 1]; --p; }
      bd[p] = d2; bi[p] = cj;
    }
  }
#pragma unroll
  for (int k = 0; k < K_NB; ++k) { idxOut[i * K_NB + k] = bi[k]; dOut[i * K_NB + k] = bd[k]; }
}

// ---- GEMM1 per node: (16 nbrs x 33pad64) @ (64 x 48) via 6 WMMAs, then
//      leaky + reduce over neighbors -> S[node][64] f16 ----
__global__ void __launch_bounds__(32) mp_gemm1(const float* selfF, const float* nodeF,
                                               const int* idx, const float* dist,
                                               const _Float16* w1s, const float* b1,
                                               _Float16* S16) {
  __shared__ __align__(16) _Float16 F[16 * 64];
  const int node = blockIdx.x;
  const int lane = threadIdx.x;
  for (int t = lane; t < 16 * 32; t += 32) {         // packed pair stores
    const int r = t >> 5, cp = t & 31, col = cp * 2;
    unsigned int u = 0u;
    if (col < 16) {
      const float* src = selfF + node * 16 + col;
      u = pack2(src[0], src[1]);
    } else if (col < 32) {
      const float* src = nodeF + idx[node * 16 + r] * 16 + (col - 16);
      u = pack2(src[0], src[1]);
    } else if (col == 32) {
      u = pack2(dist[node * 16 + r], 0.f);
    }
    *(unsigned int*)(F + r * 64 + col) = u;
  }
  __syncthreads();
  const int col0 = lane & 15, hh = lane >> 4;
#pragma unroll
  for (int nT = 0; nT < 3; ++nT) {
    v8f c = {};
#pragma unroll
    for (int kS = 0; kS < 2; ++kS) {
      v16h a = load_a_frag(F, 64, kS * 32);
      v16h b = load_b_fragS(w1s, kS * 3 + nT);
      c = __builtin_amdgcn_wmma_f32_16x16x32_f16(false, a, false, b, (short)0, c, false, false);
    }
    const int col = nT * 16 + col0;
    const float bias = (col < H) ? b1[col] : 0.f;
    float s = 0.f;
#pragma unroll
    for (int r = 0; r < 8; ++r) s += leaky(c[r] + bias);   // 8 neighbor rows
    s += __shfl_xor(s, 16);                                 // + other 8 rows
    if (hh == 0) S16[node * 64 + col] = (_Float16)s;
  }
  if (lane < 8) *(unsigned int*)(S16 + node * 64 + 48 + 2 * lane) = 0u;
}

// ---- GEMM2: 16 nodes/wave: S(16x64) @ w2(64x16) (2 WMMAs) + 16*b2,
//      groupnorm(2x8) + leaky + residual accumulate ----
__global__ void __launch_bounds__(32) mp_gemm2(const _Float16* S16, const _Float16* w2s,
                                               const float* b2, const float* gw,
                                               const float* gb, float* inout) {
  __shared__ float Dls[16 * 16];
  const int tile = blockIdx.x;
  const int lane = threadIdx.x;
  v8f c = {};
#pragma unroll
  for (int kS = 0; kS < 2; ++kS) {
    v16h a = load_a_frag(S16 + tile * 16 * 64, 64, kS * 32);
    v16h b = load_b_fragS(w2s, kS);
    c = __builtin_amdgcn_wmma_f32_16x16x32_f16(false, a, false, b, (short)0, c, false, false);
  }
  const int col = lane & 15, hh = lane >> 4;
  const float bias = 16.f * b2[col];   // sum_k (h@w2 + b2) = (sum_k h)@w2 + 16*b2
#pragma unroll
  for (int r = 0; r < 8; ++r) Dls[(r + 8 * hh) * 16 + col] = c[r] + bias;
  __syncthreads();
  if (lane < 16) {
    const int row = lane;
    const float* x = &Dls[row * 16];
#pragma unroll
    for (int g = 0; g < 2; ++g) {
      float mu = 0.f, m2 = 0.f;
#pragma unroll
      for (int t = 0; t < 8; ++t) { const float v = x[g * 8 + t]; mu += v; m2 += v * v; }
      mu *= 0.125f; m2 *= 0.125f;
      const float inv = rsqrtf((m2 - mu * mu) + EPSV);
#pragma unroll
      for (int t = 0; t < 8; ++t) {
        const int cc = g * 8 + t;
        const float y = (x[cc] - mu) * inv * gw[cc] + gb[cc];
        inout[(tile * 16 + row) * 16 + cc] += leaky(y);
      }
    }
  }
}

__global__ void init_ones(float* p, int n) {
  const int i = blockIdx.x * blockDim.x + threadIdx.x;
  if (i < n) p[i] = 1.0f;
}

extern "C" void kernel_launch(void* const* d_in, const int* in_sizes, int n_in,
                              void* d_out, int out_size, void* d_ws, size_t ws_size,
                              hipStream_t stream) {
  (void)out_size; (void)ws_size;
  const float* xyz       = (const float*)d_in[0];
  const float* atom_xyz  = (const float*)d_in[1];
  const float* atomtypes = (const float*)d_in[2];
  // d_in[3]=batch, d_in[4]=atom_batch: structurally repeat(arange(6), n/6) — derived arithmetically.

  const float *tt_w1, *tt_b1, *tt_w2, *tt_b2;
  const float *aa_w1, *aa_b1, *aa_w2, *aa_b2, *aa_gw, *aa_gb;
  const float *se_w1, *se_b1, *se_w2, *se_b2, *se_gw, *se_gb;
  if (n_in >= 21 && in_sizes[5] == 256) {            // dict insertion order
    tt_w1=(const float*)d_in[5];  tt_b1=(const float*)d_in[6];
    tt_w2=(const float*)d_in[7];  tt_b2=(const float*)d_in[8];
    aa_w1=(const float*)d_in[9];  aa_b1=(const float*)d_in[10];
    aa_w2=(const float*)d_in[11]; aa_b2=(const float*)d_in[12];
    aa_gw=(const float*)d_in[13]; aa_gb=(const float*)d_in[14];
    se_w1=(const float*)d_in[15]; se_b1=(const float*)d_in[16];
    se_w2=(const float*)d_in[17]; se_b2=(const float*)d_in[18];
    se_gw=(const float*)d_in[19]; se_gb=(const float*)d_in[20];
  } else {                                           // jax sorted-pytree order
    aa_b1=(const float*)d_in[5];  aa_b2=(const float*)d_in[6];
    aa_gb=(const float*)d_in[7];  aa_gw=(const float*)d_in[8];
    aa_w1=(const float*)d_in[9];  aa_w2=(const float*)d_in[10];
    se_b1=(const float*)d_in[11]; se_b2=(const float*)d_in[12];
    se_gb=(const float*)d_in[13]; se_gw=(const float*)d_in[14];
    se_w1=(const float*)d_in[15]; se_w2=(const float*)d_in[16];
    tt_b1=(const float*)d_in[17]; tt_b2=(const float*)d_in[18];
    tt_w1=(const float*)d_in[19]; tt_w2=(const float*)d_in[20];
  }

  char* wsp = (char*)d_ws;
  auto take = [&](size_t bytes) { char* p = wsp; wsp += (bytes + 255) & ~(size_t)255; return p; };
  float*    outA = (float*)take((size_t)N_ATOM * 16 * 4);
  int*      idxA = (int*)take((size_t)N_ATOM * 16 * 4);
  float*    dA   = (float*)take((size_t)N_ATOM * 16 * 4);
  int*      idxS = (int*)take((size_t)N_SURF * 16 * 4);
  float*    dS   = (float*)take((size_t)N_SURF * 16 * 4);
  _Float16* S16  = (_Float16*)take((size_t)N_SURF * 64 * 2);
  _Float16* w1s  = (_Float16*)take((size_t)6 * 6 * 512 * 2);
  _Float16* w2s  = (_Float16*)take((size_t)6 * 2 * 512 * 2);
  _Float16* tt1s = (_Float16*)take((size_t)512 * 2);
  _Float16* tt2s = (_Float16*)take((size_t)512 * 2);

  prep_weights<<<8, 256, 0, stream>>>(tt_w1, tt_w2, aa_w1, se_w1, aa_w2, se_w2,
                                      tt1s, tt2s, w1s, w2s);
  tt_kernel<<<N_ATOM / 16, 32, 0, stream>>>(atomtypes, tt_b1, tt_b2, tt1s, tt2s, outA);
  knn_kernel<<<(N_ATOM + 127) / 128, 128, 0, stream>>>(atom_xyz, atom_xyz, N_ATOM, APM, APM, 1, idxA, dA);
  knn_kernel<<<(N_SURF + 127) / 128, 128, 0, stream>>>(xyz, atom_xyz, N_SURF, SPM, APM, 0, idxS, dS);

  for (int l = 0; l < 3; ++l) {
    mp_gemm1<<<N_ATOM, 32, 0, stream>>>(outA, outA, idxA, dA,
                                        w1s + l * 6 * 512, aa_b1 + l * H, S16);
    mp_gemm2<<<N_ATOM / 16, 32, 0, stream>>>(S16, w2s + l * 2 * 512, aa_b2 + l * D,
                                             aa_gw + l * D, aa_gb + l * D, outA);
  }

  float* emb = (float*)d_out;
  init_ones<<<(N_SURF * D + 255) / 256, 256, 0, stream>>>(emb, N_SURF * D);
  for (int l = 0; l < 3; ++l) {
    mp_gemm1<<<N_SURF, 32, 0, stream>>>(emb, outA, idxS, dS,
                                        w1s + (3 + l) * 6 * 512, se_b1 + l * H, S16);
    mp_gemm2<<<N_SURF / 16, 32, 0, stream>>>(S16, w2s + (3 + l) * 2 * 512, se_b2 + l * D,
                                             se_gw + l * D, se_gb + l * D, emb);
  }
}
